// NTM_16518444220518
// MI455X (gfx1250) — compile-verified, matrix operand order checked by reading, not run on hardware
//
#include <hip/hip_runtime.h>
#include <hip/hip_bf16.h>

// NTM forward for MI455X (gfx1250, wave32, WMMA).
// B=64,S=256,V=256,D=128,C=256,N=128,M=32,H=4
#define BB 64
#define SS 256
#define VV 256
#define DD 128
#define CC 256
#define NN 128
#define MMW 32
#define HHN 4

typedef __attribute__((ext_vector_type(16))) _Float16 v16h;
typedef __attribute__((ext_vector_type(8)))  float    v8f;
typedef __attribute__((ext_vector_type(8)))  unsigned v8u;

// K index for element i (0..15) of a 16-bit WMMA A/B operand, half = lane>>4
// (ISA 7.12.2: lanes 0-15 hold K 0..7 / 16..23, lanes 16-31 hold K 8..15 / 24..31)
__device__ __forceinline__ int kmap16(int i, int half) {
  return i + 8 * half + ((i >= 8) ? 8 : 0);
}

__device__ __forceinline__ float sigmf(float x)     { return 1.f / (1.f + __expf(-x)); }
__device__ __forceinline__ float softplusf(float x) { return (x > 20.f) ? x : __logf(1.f + __expf(x)); }
__device__ __forceinline__ float wave_max(float v)  {
  #pragma unroll
  for (int o = 16; o > 0; o >>= 1) v = fmaxf(v, __shfl_xor(v, o, 32));
  return v;
}
__device__ __forceinline__ float wave_sum(float v)  {
  #pragma unroll
  for (int o = 16; o > 0; o >>= 1) v += __shfl_xor(v, o, 32);
  return v;
}

// ---------------- weight packing (run once per launch) ----------------
// Wcat[512,1024] = [W_ih^T ; W_hh^T] packed as B-fragments:
// frag index ((nt*16+kt)*32+lane)*16+i ; n = nt*16+(lane&15), k = kt*32+kmap16(i,lane>>4)
__global__ void pack_wcat_k(const float* __restrict__ W_ih, const float* __restrict__ W_hh,
                            _Float16* __restrict__ wcat) {
  int idx = blockIdx.x * blockDim.x + threadIdx.x;
  if (idx >= 64 * 16 * 32 * 16) return;
  int i = idx & 15, lane = (idx >> 4) & 31, kt = (idx >> 9) & 15, nt = idx >> 13;
  int n = nt * 16 + (lane & 15);
  int k = kt * 32 + kmap16(i, lane >> 4);
  float v = (k < 256) ? W_ih[n * 256 + k] : W_hh[n * 256 + (k - 256)];
  wcat[idx] = (_Float16)v;
}

// Whead[256, 448]: per head hh (112 cols): key(32) erase(32) add(32) beta gate gamma shift(3) pad(10)
__global__ void pack_whead_k(const float* __restrict__ key_W, const float* __restrict__ erase_W,
                             const float* __restrict__ add_W, const float* __restrict__ beta_W,
                             const float* __restrict__ gate_W, const float* __restrict__ gamma_W,
                             const float* __restrict__ shift_W, _Float16* __restrict__ whead) {
  int idx = blockIdx.x * blockDim.x + threadIdx.x;
  if (idx >= 28 * 8 * 32 * 16) return;
  int i = idx & 15, lane = (idx >> 4) & 31, kt = (idx >> 9) & 7, nt = idx >> 12;
  int n  = nt * 16 + (lane & 15);
  int k  = kt * 32 + kmap16(i, lane >> 4);
  int hh = n / 112, cc = n % 112;
  float v = 0.f;
  if (cc < 32)       v = key_W  [(hh * 256 + k) * 32 + cc];
  else if (cc < 64)  v = erase_W[(hh * 256 + k) * 32 + (cc - 32)];
  else if (cc < 96)  v = add_W  [(hh * 256 + k) * 32 + (cc - 64)];
  else if (cc == 96) v = beta_W [hh * 256 + k];
  else if (cc == 97) v = gate_W [hh * 256 + k];
  else if (cc == 98) v = gamma_W[hh * 256 + k];
  else if (cc < 102) v = shift_W[(hh * 256 + k) * 3 + (cc - 99)];
  whead[idx] = (_Float16)v;
}

// Wout[384,256]: logits = [h, read] @ out_W
__global__ void pack_wout_k(const float* __restrict__ out_W, _Float16* __restrict__ wout) {
  int idx = blockIdx.x * blockDim.x + threadIdx.x;
  if (idx >= 16 * 12 * 32 * 16) return;
  int i = idx & 15, lane = (idx >> 4) & 31;
  int rem = idx >> 9;            // nt*12+kt
  int kt = rem % 12, nt = rem / 12;
  int n = nt * 16 + (lane & 15);
  int k = kt * 32 + kmap16(i, lane >> 4);
  wout[idx] = (_Float16)out_W[k * 256 + n];
}

__global__ void copy_mem_k(const float* __restrict__ src, float* __restrict__ dst) {
  int idx = blockIdx.x * blockDim.x + threadIdx.x;
  if (idx < BB * NN * MMW) dst[idx] = src[idx];
}

// ---------------- main persistent NTM kernel ----------------
__global__ __launch_bounds__(512) void ntm_main_k(
    const int*   __restrict__ x,
    const float* __restrict__ embedding,
    const _Float16* __restrict__ wcat,
    const _Float16* __restrict__ whead,
    const _Float16* __restrict__ wout,
    const float* __restrict__ b_ih,   const float* __restrict__ b_hh,
    const float* __restrict__ key_b,  const float* __restrict__ beta_b,
    const float* __restrict__ gate_b, const float* __restrict__ shift_b,
    const float* __restrict__ gamma_b,const float* __restrict__ erase_b,
    const float* __restrict__ add_b,  const float* __restrict__ out_b,
    float* __restrict__ memws,
    float* __restrict__ out) {
  // LDS state for 16 batch rows (~156 KB, WGP has 320 KB)
  __shared__ float    s_c [16][CC];          // LSTM cell state
  __shared__ _Float16 s_a [16][512];         // A staging: [emb(128) | rv(128) | h(256)] as f16
  __shared__ float    s_rw[16][HHN][NN];     // previous read weights
  __shared__ float    s_ph[16][448];         // head projections
  __shared__ float    s_w [16][HHN][NN];     // sim -> weights
  __shared__ float    s_kn[16][HHN][MMW];    // normalized keys
  __shared__ float    s_beta[16][HHN], s_gate[16][HHN], s_gamma[16][HHN];
  __shared__ float    s_shift[16][HHN][3];
  __shared__ float    s_er[16][HHN][MMW], s_ad[16][HHN][MMW];
  __shared__ float    s_bcat[1024];          // b_ih + b_hh
  __shared__ float    s_hb[448];             // head biases in packed column order
  __shared__ float    s_ob[VV];              // out_b

  const int tid   = threadIdx.x;
  const int lane  = tid & 31;
  const int wv    = tid >> 5;      // 0..15
  const int half  = lane >> 4;
  const int col16 = lane & 15;
  const int bb    = blockIdx.x;    // 0..3  (16 batch rows each)
  float* mem = memws + (size_t)bb * 16 * NN * MMW;

  // ---- init ----
  for (int j = tid; j < 16 * CC; j += 512)        ((float*)s_c)[j] = 0.f;
  for (int j = tid; j < 16 * 512; j += 512)       ((_Float16*)s_a)[j] = (_Float16)0.f;
  for (int j = tid; j < 16 * HHN * NN; j += 512)  ((float*)s_rw)[j] = 0.f;
  for (int j = tid; j < 1024; j += 512)           s_bcat[j] = b_ih[j] + b_hh[j];
  if (tid < 448) {
    int n = tid, hh = n / 112, cc = n % 112; float bv = 0.f;
    if (cc < 32)       bv = key_b  [hh * 32 + cc];
    else if (cc < 64)  bv = erase_b[hh * 32 + cc - 32];
    else if (cc < 96)  bv = add_b  [hh * 32 + cc - 64];
    else if (cc == 96) bv = beta_b [hh];
    else if (cc == 97) bv = gate_b [hh];
    else if (cc == 98) bv = gamma_b[hh];
    else if (cc < 102) bv = shift_b[hh * 3 + cc - 99];
    s_hb[tid] = bv;
  }
  if (tid < VV) s_ob[tid] = out_b[tid];
  __syncthreads();

  // A-fragment assembly from LDS staging (pairs of f16 -> 8 dword loads)
  auto load_af = [&](int mrow, int colbase) -> v16h {
    v8u tacc;
    #pragma unroll
    for (int j = 0; j < 8; ++j) {
      int k = colbase + 2 * j + 8 * half + ((j >= 4) ? 8 : 0);
      tacc[j] = *(const unsigned*)&s_a[mrow][k];
    }
    return __builtin_bit_cast(v16h, tacc);
  };

  for (int t = 0; t < SS; ++t) {
    // -- phase 1: embedding lookup -> s_a[:,0:128) (f16)
    for (int it = 0; it < 4; ++it) {
      int task = tid + it * 512;                // (m, d)
      int m = task >> 7, d = task & 127;
      int tok = x[(bb * 16 + m) * SS + t];
      s_a[m][d] = (_Float16)embedding[tok * DD + d];
    }
    __syncthreads();

    // -- phase 2: fused LSTM GEMM g = [li|h][16,512] @ Wcat[512,1024]
    // wave wv owns gate columns 16wv..16wv+15 for blocks i,f,g,o (tiles wv,wv+16,wv+32,wv+48)
    v8f acc0 = {}, acc1 = {}, acc2 = {}, acc3 = {};
    for (int kt = 0; kt < 16; ++kt) {
      v16h af = load_af(col16, kt * 32);
      v16h b0 = *((const v16h*)wcat + (((wv     ) * 16 + kt) * 32 + lane));
      v16h b1 = *((const v16h*)wcat + (((wv + 16) * 16 + kt) * 32 + lane));
      v16h b2 = *((const v16h*)wcat + (((wv + 32) * 16 + kt) * 32 + lane));
      v16h b3 = *((const v16h*)wcat + (((wv + 48) * 16 + kt) * 32 + lane));
      acc0 = __builtin_amdgcn_wmma_f32_16x16x32_f16(false, af, false, b0, (short)0, acc0, false, false);
      acc1 = __builtin_amdgcn_wmma_f32_16x16x32_f16(false, af, false, b1, (short)0, acc1, false, false);
      acc2 = __builtin_amdgcn_wmma_f32_16x16x32_f16(false, af, false, b2, (short)0, acc2, false, false);
      acc3 = __builtin_amdgcn_wmma_f32_16x16x32_f16(false, af, false, b3, (short)0, acc3, false, false);
    }
    __syncthreads();   // all s_a reads done before h is overwritten

    // -- phase 3: register-resident LSTM cell update; write h (f16) into s_a[:,256:512)
    {
      int col = wv * 16 + col16;
      float bi = s_bcat[col], bf2 = s_bcat[256 + col], bg = s_bcat[512 + col], bo = s_bcat[768 + col];
      #pragma unroll
      for (int r = 0; r < 8; ++r) {
        int mrow = r + 8 * half;
        float iv = acc0[r] + bi;
        float fv = acc1[r] + bf2;
        float gv = acc2[r] + bg;
        float ov = acc3[r] + bo;
        float cold = s_c[mrow][col];
        float cn = sigmf(fv) * cold + sigmf(iv) * tanhf(gv);
        float hn = sigmf(ov) * tanhf(cn);
        s_c[mrow][col] = cn;
        s_a[mrow][256 + col] = (_Float16)hn;
      }
    }
    __syncthreads();

    // -- phase 4: head projections P = h[16,256] @ Whead[256,448]
    for (int nt = wv; nt < 28; nt += 16) {
      v8f acc = {};
      for (int kt = 0; kt < 8; ++kt) {
        v16h af = load_af(col16, 256 + kt * 32);
        v16h bf = *((const v16h*)whead + ((nt * 8 + kt) * 32 + lane));
        acc = __builtin_amdgcn_wmma_f32_16x16x32_f16(false, af, false, bf, (short)0, acc, false, false);
      }
      int n = nt * 16 + col16;
      float hb = s_hb[n];
      #pragma unroll
      for (int r = 0; r < 8; ++r) s_ph[r + 8 * half][n] = acc[r] + hb;
    }
    __syncthreads();

    // -- phase 5: head nonlinearities + key normalization
    if (tid < 64) {
      int m = tid >> 2, hh = tid & 3, base = hh * 112;
      float kv[32]; float nrm = 0.f;
      #pragma unroll
      for (int j = 0; j < 32; ++j) { kv[j] = s_ph[m][base + j]; nrm += kv[j] * kv[j]; }
      float inv = 1.f / fmaxf(sqrtf(nrm), 1e-12f);
      #pragma unroll
      for (int j = 0; j < 32; ++j) s_kn[m][hh][j] = kv[j] * inv;
      #pragma unroll
      for (int j = 0; j < 32; ++j) {
        s_er[m][hh][j] = sigmf(s_ph[m][base + 32 + j]);
        s_ad[m][hh][j] = tanhf(s_ph[m][base + 64 + j]);
      }
      s_beta [m][hh] = softplusf(s_ph[m][base + 96]);
      s_gate [m][hh] = sigmf(s_ph[m][base + 97]);
      s_gamma[m][hh] = 1.f + softplusf(s_ph[m][base + 98]);
      float a0 = s_ph[m][base + 99], a1 = s_ph[m][base + 100], a2 = s_ph[m][base + 101];
      float mx = fmaxf(a0, fmaxf(a1, a2));
      float e0 = __expf(a0 - mx), e1 = __expf(a1 - mx), e2 = __expf(a2 - mx);
      float si = 1.f / (e0 + e1 + e2);
      s_shift[m][hh][0] = e0 * si; s_shift[m][hh][1] = e1 * si; s_shift[m][hh][2] = e2 * si;
    }
    __syncthreads();

    // -- phase 6: cosine similarity sim[m][hh][n]
    for (int it = 0; it < 4; ++it) {
      int task = tid + it * 512;                // (m, n)
      int m = task >> 7, n = task & 127;
      const float4* mp = (const float4*)(mem + (size_t)(m * NN + n) * MMW);
      float mv[32]; float nrm = 0.f;
      #pragma unroll
      for (int j = 0; j < 8; ++j) {
        float4 v4 = mp[j];
        mv[4 * j + 0] = v4.x; mv[4 * j + 1] = v4.y; mv[4 * j + 2] = v4.z; mv[4 * j + 3] = v4.w;
      }
      #pragma unroll
      for (int j = 0; j < 32; ++j) nrm += mv[j] * mv[j];
      float inv = 1.f / fmaxf(sqrtf(nrm), 1e-12f);
      #pragma unroll
      for (int hh = 0; hh < 4; ++hh) {
        float d = 0.f;
        #pragma unroll
        for (int j = 0; j < 32; ++j) d += mv[j] * s_kn[m][hh][j];
        s_w[m][hh][n] = d * inv;
      }
    }
    __syncthreads();

    // -- phase 7: addressing (softmax, gate blend, circular shift, sharpen); wave wv owns row wv
    {
      int m = wv;
      for (int hh = 0; hh < 4; ++hh) {
        float beta = s_beta[m][hh], gate = s_gate[m][hh], gamma = s_gamma[m][hh];
        float sh0 = s_shift[m][hh][0], sh1 = s_shift[m][hh][1], sh2 = s_shift[m][hh][2];
        float v[4]; float mx = -1e30f;
        #pragma unroll
        for (int q = 0; q < 4; ++q) { v[q] = beta * s_w[m][hh][q * 32 + lane]; mx = fmaxf(mx, v[q]); }
        mx = wave_max(mx);
        float sum = 0.f;
        #pragma unroll
        for (int q = 0; q < 4; ++q) { v[q] = __expf(v[q] - mx); sum += v[q]; }
        float isum = 1.f / wave_sum(sum);
        #pragma unroll
        for (int q = 0; q < 4; ++q) {
          int n = q * 32 + lane;
          s_w[m][hh][n] = gate * (v[q] * isum) + (1.f - gate) * s_rw[m][hh][n];
        }
        __builtin_amdgcn_wave_barrier();
        float wp[4]; float psum = 0.f;
        #pragma unroll
        for (int q = 0; q < 4; ++q) {
          int n = q * 32 + lane;
          float ws = sh0 * s_w[m][hh][(n + 1) & 127] + sh1 * s_w[m][hh][n] + sh2 * s_w[m][hh][(n - 1) & 127];
          wp[q] = __powf(fmaxf(ws, 0.f), gamma);
          psum += wp[q];
        }
        float ipn = 1.f / wave_sum(psum);
        __builtin_amdgcn_wave_barrier();
        #pragma unroll
        for (int q = 0; q < 4; ++q) {
          int n = q * 32 + lane;
          float wf = wp[q] * ipn;
          s_w[m][hh][n] = wf;
          s_rw[m][hh][n] = wf;
        }
      }
    }
    __syncthreads();

    // -- phase 8: read vectors (before write), staged as f16 into s_a[:,128:256) for next step + logits
    for (int it = 0; it < 4; ++it) {
      int task = tid + it * 512;                // (m, hh, mm)
      int m = task >> 7, rest = task & 127, hh = rest >> 5, mm = rest & 31;
      const float* mp = mem + (size_t)m * NN * MMW + mm;
      float acc = 0.f;
      for (int n = 0; n < NN; ++n) acc += s_w[m][hh][n] * mp[(size_t)n * MMW];
      s_a[m][128 + hh * MMW + mm] = (_Float16)acc;
    }
    __syncthreads();

    // -- phase 9: memory erase/add update
    for (int it = 0; it < 4; ++it) {
      int task = tid + it * 512;                // (m, n)
      int m = task >> 7, n = task & 127;
      float w0 = s_w[m][0][n], w1 = s_w[m][1][n], w2 = s_w[m][2][n], w3 = s_w[m][3][n];
      float* mp = mem + (size_t)(m * NN + n) * MMW;
      #pragma unroll
      for (int mm = 0; mm < 32; ++mm) {
        float E = w0 * s_er[m][0][mm] + w1 * s_er[m][1][mm] + w2 * s_er[m][2][mm] + w3 * s_er[m][3][mm];
        float A = w0 * s_ad[m][0][mm] + w1 * s_ad[m][1][mm] + w2 * s_ad[m][2][mm] + w3 * s_ad[m][3][mm];
        mp[mm] = mp[mm] * (1.f - E) + A;
      }
    }
    __syncthreads();

    // -- phase 10: logits = [h(256) | read(128)] @ Wout[384,256] + out_b ; wave wv -> cols 16wv..16wv+15
    {
      v8f acc = {};
      #pragma unroll
      for (int kk = 0; kk < 12; ++kk) {
        int colbase = (kk < 8) ? (256 + kk * 32) : (128 + (kk - 8) * 32);
        v16h af = load_af(col16, colbase);
        v16h bf = *((const v16h*)wout + ((wv * 12 + kk) * 32 + lane));
        acc = __builtin_amdgcn_wmma_f32_16x16x32_f16(false, af, false, bf, (short)0, acc, false, false);
      }
      int n = wv * 16 + col16;
      float ob = s_ob[n];
      #pragma unroll
      for (int r = 0; r < 8; ++r) {
        int mrow = r + 8 * half;
        out[(((size_t)(bb * 16 + mrow)) * SS + t) * VV + n] = acc[r] + ob;
      }
    }
    __syncthreads();
  }
}

extern "C" void kernel_launch(void* const* d_in, const int* in_sizes, int n_in,
                              void* d_out, int out_size, void* d_ws, size_t ws_size,
                              hipStream_t stream) {
  (void)in_sizes; (void)n_in; (void)out_size; (void)ws_size;
  const int*   x         = (const int*)  d_in[0];
  const float* memory0   = (const float*)d_in[1];
  const float* embedding = (const float*)d_in[2];
  const float* W_ih      = (const float*)d_in[3];
  const float* b_ih      = (const float*)d_in[4];
  const float* W_hh      = (const float*)d_in[5];
  const float* b_hh      = (const float*)d_in[6];
  const float* key_W     = (const float*)d_in[7];
  const float* key_b     = (const float*)d_in[8];
  const float* beta_W    = (const float*)d_in[9];
  const float* beta_b    = (const float*)d_in[10];
  const float* gate_W    = (const float*)d_in[11];
  const float* gate_b    = (const float*)d_in[12];
  const float* shift_W   = (const float*)d_in[13];
  const float* shift_b   = (const float*)d_in[14];
  const float* gamma_W   = (const float*)d_in[15];
  const float* gamma_b   = (const float*)d_in[16];
  const float* erase_W   = (const float*)d_in[17];
  const float* erase_b   = (const float*)d_in[18];
  const float* add_W     = (const float*)d_in[19];
  const float* add_b     = (const float*)d_in[20];
  const float* out_W     = (const float*)d_in[21];
  const float* out_b     = (const float*)d_in[22];

  char* ws = (char*)d_ws;
  _Float16* wcat  = (_Float16*)(ws);                               // 1 MB
  _Float16* whead = (_Float16*)(ws + 1048576);                     // 224 KB
  _Float16* wout  = (_Float16*)(ws + 1048576 + 229376);            // 192 KB
  float*    memws = (float*)   (ws + 1048576 + 229376 + 196608);   // 1 MB

  pack_wcat_k <<<2048, 256, 0, stream>>>(W_ih, W_hh, wcat);
  pack_whead_k<<<448,  256, 0, stream>>>(key_W, erase_W, add_W, beta_W, gate_W, gamma_W, shift_W, whead);
  pack_wout_k <<<384,  256, 0, stream>>>(out_W, wout);
  copy_mem_k  <<<1024, 256, 0, stream>>>(memory0, memws);

  ntm_main_k<<<4, 512, 0, stream>>>(x, embedding, wcat, whead, wout,
                                    b_ih, b_hh, key_b, beta_b, gate_b, shift_b,
                                    gamma_b, erase_b, add_b, out_b,
                                    memws, (float*)d_out);
}
